// MultiHeadAttention_42949672960187
// MI455X (gfx1250) — compile-verified, matrix operand order checked by reading, not run on hardware
//
#include <hip/hip_runtime.h>

// ---------------------------------------------------------------------------
// MHA forward for B=4, S=2048, E=1024, H=16, D=64 on gfx1250 (MI455X).
// Matmuls via v_wmma_f32_16x16x32_bf16; LDS tile staging via async-to-LDS
// (ASYNCcnt) when available; flash-attention streaming (no S x S scores).
// ---------------------------------------------------------------------------

typedef __bf16 bf16;
typedef __attribute__((ext_vector_type(16))) __bf16 v16bf;
typedef __attribute__((ext_vector_type(8)))  __bf16 v8bf;
typedef __attribute__((ext_vector_type(8)))  float  v8f;
typedef __attribute__((ext_vector_type(4)))  int    v4i;

static constexpr int Bb = 4;
static constexpr int Ss = 2048;
static constexpr int Ee = 1024;
static constexpr int Hh = 16;
// D = 64 per head.

#if __has_builtin(__builtin_amdgcn_global_load_async_to_lds_b128)
#define HAVE_ASYNC_LDS 1
typedef __attribute__((address_space(1))) v4i gas_v4i;
typedef __attribute__((address_space(3))) v4i las_v4i;
#endif

// 16-byte global -> LDS copy. Async (ASYNCcnt) when the gfx1250 builtin
// exists; otherwise synchronous load+ds_store (DScnt).
__device__ __forceinline__ void copy16_to_lds(bf16* lptr, const bf16* gptr) {
#ifdef HAVE_ASYNC_LDS
  __builtin_amdgcn_global_load_async_to_lds_b128(
      (gas_v4i*)(void*)(const_cast<bf16*>(gptr)),
      (las_v4i*)(void*)lptr, 0, 0);
#else
  *(v8bf*)lptr = *(const v8bf*)gptr;
#endif
}

// Make this wave's pending LDS-fill visible before the block barrier.
__device__ __forceinline__ void wait_lds_fill() {
#ifdef HAVE_ASYNC_LDS
#if __has_builtin(__builtin_amdgcn_s_wait_asynccnt)
  __builtin_amdgcn_s_wait_asynccnt(0);
#else
  asm volatile("s_wait_asynccnt 0" ::: "memory");
#endif
#else
  asm volatile("s_wait_dscnt 0" ::: "memory");
#endif
}

__device__ __forceinline__ v8f wmma_bf16(v16bf a, v16bf b, v8f c) {
  // (neg_a, A, neg_b, B, c_mod, C, reuse_a, reuse_b)
  return __builtin_amdgcn_wmma_f32_16x16x32_bf16(false, a, false, b, (short)0, c,
                                                 false, false);
}

// Load one 16x32 bf16 A/B fragment (v16bf, 8 VGPRs) from a row-major matrix
// (global or LDS). ISA layout: lanes 0-15 hold K = kb+[0..7] then kb+[16..23];
// lanes 16-31 hold K = kb+[8..15] then kb+[24..31]. Two 16B chunks per lane.
__device__ __forceinline__ v16bf load_frag(const bf16* base, int ld, int row,
                                           int k0, int lane) {
  const bf16* p = base + (size_t)row * ld + k0 + ((lane >> 4) << 3);
  v8bf lo = *(const v8bf*)(p);
  v8bf hi = *(const v8bf*)(p + 16);
  return __builtin_shufflevector(lo, hi, 0, 1, 2, 3, 4, 5, 6, 7,
                                         8, 9, 10, 11, 12, 13, 14, 15);
}

// 16-lane reductions (xor masks 1,2,4,8 stay inside each half-wave) using
// ds_swizzle_b32 group-of-32 immediates: xor_mask<<10 | and_mask(0x1f).
#define SWZ_XOR(x, m)                                                        \
  __int_as_float(__builtin_amdgcn_ds_swizzle(__float_as_int(x),              \
                                             ((m) << 10) | 0x1f))
__device__ __forceinline__ float redmax16(float v) {
  v = fmaxf(v, SWZ_XOR(v, 1));
  v = fmaxf(v, SWZ_XOR(v, 2));
  v = fmaxf(v, SWZ_XOR(v, 4));
  v = fmaxf(v, SWZ_XOR(v, 8));
  return v;
}
__device__ __forceinline__ float redsum16(float v) {
  v += SWZ_XOR(v, 1);
  v += SWZ_XOR(v, 2);
  v += SWZ_XOR(v, 4);
  v += SWZ_XOR(v, 8);
  return v;
}

// ---------------------------------------------------------------------------
// fp32 -> bf16 conversion, 8 elements/thread
// ---------------------------------------------------------------------------
__global__ void f32_to_bf16_kernel(const float* __restrict__ in,
                                   bf16* __restrict__ out, int n8) {
  int i = blockIdx.x * blockDim.x + threadIdx.x;
  if (i < n8) {
    const float4 a = ((const float4*)in)[i * 2 + 0];
    const float4 b = ((const float4*)in)[i * 2 + 1];
    v8bf o;
    o[0] = (bf16)a.x; o[1] = (bf16)a.y; o[2] = (bf16)a.z; o[3] = (bf16)a.w;
    o[4] = (bf16)b.x; o[5] = (bf16)b.y; o[6] = (bf16)b.z; o[7] = (bf16)b.w;
    *(v8bf*)(out + (size_t)i * 8) = o;
  }
}

// ---------------------------------------------------------------------------
// GEMM: C[M,N] = A[M,K](bf16 rm) * W[N,K](bf16 rm)^T + bias[N]
// Block 256 thr = 8 waves (4M x 2N of 32x32), block tile 128x64, K-step 32.
// A/W K-slabs double-buffered in LDS via async copies.
// MODE 0: QKV epilogue (Q,K -> qk[M,2048]; V -> vT transposed).
// MODE 1: f32 output.
// ---------------------------------------------------------------------------
template <int MODE>
__global__ __launch_bounds__(256) void gemm_bf16_wmma(
    const bf16* __restrict__ A, const bf16* __restrict__ W,
    const float* __restrict__ bias, float* __restrict__ Cf,
    bf16* __restrict__ Cqk, bf16* __restrict__ vT, int M, int N, int K) {
  __shared__ __align__(16) bf16 Asm[2][128 * 40];  // 128 rows, ld 40 (pad 8)
  __shared__ __align__(16) bf16 Wsm[2][64 * 40];

  const int tid = threadIdx.x;
  const int lane = tid & 31;
  const int wave = tid >> 5;
  const int wm = wave >> 1, wn = wave & 1;
  const int bm = blockIdx.y * 128;
  const int bn = blockIdx.x * 64;

  // Stage one K-slab: A 128x32 (2 xfers/thread), W 64x32 (1 xfer/thread).
  auto stage = [&](int buf, int kb) {
    const int c = tid & 3;   // 8-element chunk within the 32-wide slab
    const int r = tid >> 2;  // 0..63
    copy16_to_lds(&Asm[buf][r * 40 + c * 8],
                  A + (size_t)(bm + r) * K + kb + c * 8);
    copy16_to_lds(&Asm[buf][(r + 64) * 40 + c * 8],
                  A + (size_t)(bm + r + 64) * K + kb + c * 8);
    copy16_to_lds(&Wsm[buf][r * 40 + c * 8],
                  W + (size_t)(bn + r) * K + kb + c * 8);
    __builtin_prefetch(A + (size_t)(bm + r) * K + kb + 256, 0, 0);
  };

  v8f acc00 = {}, acc01 = {}, acc10 = {}, acc11 = {};
  const int nk = K / 32;
  stage(0, 0);
  for (int i = 0; i < nk; ++i) {
    wait_lds_fill();
    __syncthreads();
    if (i + 1 < nk) stage((i + 1) & 1, (i + 1) * 32);

    const bf16* At = &Asm[i & 1][0];
    const bf16* Wt = &Wsm[i & 1][0];
    v16bf a0 = load_frag(At, 40, wm * 32 + (lane & 15), 0, lane);
    v16bf a1 = load_frag(At, 40, wm * 32 + 16 + (lane & 15), 0, lane);
    v16bf b0 = load_frag(Wt, 40, wn * 32 + (lane & 15), 0, lane);
    v16bf b1 = load_frag(Wt, 40, wn * 32 + 16 + (lane & 15), 0, lane);
    acc00 = wmma_bf16(a0, b0, acc00);
    acc01 = wmma_bf16(a0, b1, acc01);
    acc10 = wmma_bf16(a1, b0, acc10);
    acc11 = wmma_bf16(a1, b1, acc11);
  }

  const int mrow = (lane >> 4) << 3;
  const int ncol = lane & 15;
  const int m0 = bm + wm * 32, n0 = bn + wn * 32;

  auto store_tile = [&](v8f acc, int mt, int nt) {
#pragma unroll
    for (int v = 0; v < 8; ++v) {
      const int m = mt + mrow + v;
      const int n = nt + ncol;
      const float val = acc[v] + bias[n];
      if (MODE == 0) {
        if (n < 2 * Ee) {
          Cqk[(size_t)m * (2 * Ee) + n] = (bf16)val;
        } else {
          const int col = n - 2 * Ee;
          const int h = col >> 6, d = col & 63;
          const int b = m >> 11, s = m & 2047;  // S = 2048
          vT[((size_t)((b * Hh + h) * 64 + d)) * Ss + s] = (bf16)val;
        }
      } else {
        Cf[(size_t)m * N + n] = val;
      }
    }
  };
  store_tile(acc00, m0, n0);
  store_tile(acc01, m0, n0 + 16);
  store_tile(acc10, m0 + 16, n0);
  store_tile(acc11, m0 + 16, n0 + 16);
}

// ---------------------------------------------------------------------------
// Flash attention. Block = 8 waves x 16 query rows = 128 q-rows of one (b,h).
// Per iteration the block stages K-tile (32x64) and V^T-tile (64x32) in LDS
// (double-buffered, async); each wave runs 8 WMMAs + online softmax.
// qk : [B*S, 2048] bf16 (Q cols 0..1023 as h*64+d, K cols 1024..2047)
// vT : [(b*H+h)*64 + d][s] bf16     ctx: [B*S, E] bf16
// ---------------------------------------------------------------------------
__global__ __launch_bounds__(256) void flash_attn_wmma(
    const bf16* __restrict__ qk, const bf16* __restrict__ vT,
    const unsigned char* __restrict__ mask, bf16* __restrict__ ctx) {
  __shared__ __align__(16) bf16 Ksm[2][32 * 72];  // 32 keys x 64 d, ld 72
  __shared__ __align__(16) bf16 Vsm[2][64 * 40];  // 64 d x 32 s, ld 40
  __shared__ __align__(16) bf16 plds[8][16 * 32]; // per-wave P staging

  const int tid = threadIdx.x;
  const int lane = tid & 31;
  const int wave = tid >> 5;
  const int bh = blockIdx.y;
  const int b = bh >> 4;  // H = 16
  const int h = bh & 15;
  const int q0 = blockIdx.x * 128 + wave * 16;
  const int qcol = h * 64;
  const int mrow = (lane >> 4) << 3;
  const int ncol = lane & 15;

  // Stage K/V tiles for key block j (256 thr: 1 xfer each per tile).
  auto stage = [&](int buf, int j) {
    {
      const int r = tid >> 3, c = tid & 7;  // K: 32 rows x 8 chunks of 8
      copy16_to_lds(&Ksm[buf][r * 72 + c * 8],
                    qk + (size_t)(b * Ss + j * 32 + r) * (2 * Ee) + Ee + qcol +
                        c * 8);
    }
    {
      const int r = tid >> 2, c = tid & 3;  // V^T: 64 rows x 4 chunks of 8
      copy16_to_lds(&Vsm[buf][r * 40 + c * 8],
                    vT + (size_t)(bh * 64 + r) * Ss + j * 32 + c * 8);
    }
  };

  // Q tile (16x64) as two K=32 A-fragments, kept in registers.
  const int rowQ = b * Ss + q0 + (lane & 15);
  const v16bf qa0 = load_frag(qk, 2 * Ee, rowQ, qcol, lane);
  const v16bf qa1 = load_frag(qk, 2 * Ee, rowQ, qcol + 32, lane);

  float mrun[8], lrun[8];
#pragma unroll
  for (int v = 0; v < 8; ++v) { mrun[v] = -1e30f; lrun[v] = 0.0f; }
  v8f o0 = {}, o1 = {}, o2 = {}, o3 = {};
  bf16* pl = &plds[wave][0];

  const int nj = Ss / 32;
  stage(0, 0);
  for (int j = 0; j < nj; ++j) {
    wait_lds_fill();
    __syncthreads();
    if (j + 1 < nj) stage((j + 1) & 1, j + 1);
    const bf16* Kt = &Ksm[j & 1][0];
    const bf16* Vt = &Vsm[j & 1][0];

    // ---- scores: two 16x16 C tiles (keys j*32+0..15 and +16..31)
    v16bf kb0lo = load_frag(Kt, 72, (lane & 15), 0, lane);
    v16bf kb0hi = load_frag(Kt, 72, (lane & 15), 32, lane);
    v16bf kb1lo = load_frag(Kt, 72, 16 + (lane & 15), 0, lane);
    v16bf kb1hi = load_frag(Kt, 72, 16 + (lane & 15), 32, lane);
    v8f s0 = {}, s1 = {};
    s0 = wmma_bf16(qa0, kb0lo, s0);
    s0 = wmma_bf16(qa1, kb0hi, s0);
    s1 = wmma_bf16(qa0, kb1lo, s1);
    s1 = wmma_bf16(qa1, kb1hi, s1);

    const float mb0 = mask[b * Ss + j * 32 + ncol]      ? 0.0f : -1e30f;
    const float mb1 = mask[b * Ss + j * 32 + 16 + ncol] ? 0.0f : -1e30f;

    // ---- online softmax (rows = mrow+v, cols across 16 lanes)
    float p0[8], p1[8], corr[8];
#pragma unroll
    for (int v = 0; v < 8; ++v) {
      const float a = s0[v] * 0.125f + mb0;  // 1/sqrt(64)
      const float c = s1[v] * 0.125f + mb1;
      const float rm = redmax16(fmaxf(a, c));
      const float mnew = fmaxf(mrun[v], rm);
      corr[v] = __expf(mrun[v] - mnew);
      p0[v] = __expf(a - mnew);
      p1[v] = __expf(c - mnew);
      const float rs = redsum16(p0[v] + p1[v]);
      lrun[v] = lrun[v] * corr[v] + rs;
      mrun[v] = mnew;
    }

    // ---- rescale O, transpose P (C-layout -> A-layout) via per-wave LDS
#pragma unroll
    for (int v = 0; v < 8; ++v) {
      o0[v] *= corr[v]; o1[v] *= corr[v]; o2[v] *= corr[v]; o3[v] *= corr[v];
      pl[(mrow + v) * 32 + ncol]      = (bf16)p0[v];
      pl[(mrow + v) * 32 + 16 + ncol] = (bf16)p1[v];
    }
    asm volatile("s_wait_dscnt 0" ::: "memory");
    const v16bf pa = load_frag(pl, 32, lane & 15, 0, lane);

    // ---- O += P(16x32) * V(32x64): V^T rows contiguous in LDS
    v16bf vb0 = load_frag(Vt, 40, 0  + (lane & 15), 0, lane);
    v16bf vb1 = load_frag(Vt, 40, 16 + (lane & 15), 0, lane);
    v16bf vb2 = load_frag(Vt, 40, 32 + (lane & 15), 0, lane);
    v16bf vb3 = load_frag(Vt, 40, 48 + (lane & 15), 0, lane);
    o0 = wmma_bf16(pa, vb0, o0);
    o1 = wmma_bf16(pa, vb1, o1);
    o2 = wmma_bf16(pa, vb2, o2);
    o3 = wmma_bf16(pa, vb3, o3);
  }

  // ---- normalize and write context (bf16, [B*S, E], col base h*64)
#pragma unroll
  for (int v = 0; v < 8; ++v) {
    const float inv = 1.0f / lrun[v];
    const size_t r = (size_t)(b * Ss + q0 + mrow + v) * Ee + qcol;
    ctx[r + 0  + ncol] = (bf16)(o0[v] * inv);
    ctx[r + 16 + ncol] = (bf16)(o1[v] * inv);
    ctx[r + 32 + ncol] = (bf16)(o2[v] * inv);
    ctx[r + 48 + ncol] = (bf16)(o3[v] * inv);
  }
}

// ---------------------------------------------------------------------------
// Launch
// ---------------------------------------------------------------------------
extern "C" void kernel_launch(void* const* d_in, const int* in_sizes, int n_in,
                              void* d_out, int out_size, void* d_ws,
                              size_t ws_size, hipStream_t stream) {
  (void)in_sizes; (void)n_in; (void)out_size; (void)ws_size;

  const float* x = (const float*)d_in[0];                     // [B,S,E]
  const unsigned char* mask = (const unsigned char*)d_in[1];  // [B,S] bool
  const float* wqkv = (const float*)d_in[2];                  // [3E,E]
  const float* bqkv = (const float*)d_in[3];                  // [3E]
  const float* wout = (const float*)d_in[4];                  // [E,E]
  const float* bout = (const float*)d_in[5];                  // [E]
  float* out = (float*)d_out;                                 // [B,S,E]

  const size_t M = (size_t)Bb * Ss;  // 8192
  bf16* x16  = (bf16*)d_ws;                       // M*E
  bf16* wq16 = x16  + M * Ee;                     // 3E*E
  bf16* wo16 = wq16 + (size_t)3 * Ee * Ee;        // E*E
  bf16* qkbf = wo16 + (size_t)Ee * Ee;            // M*2E (Q,K)
  bf16* vT   = qkbf + M * 2 * Ee;                 // (B*H*64)*S
  bf16* ctx  = vT   + (size_t)Bb * Hh * 64 * Ss;  // M*E

  {
    int n8 = (int)(M * Ee / 8);
    f32_to_bf16_kernel<<<(n8 + 255) / 256, 256, 0, stream>>>(x, x16, n8);
  }
  {
    int n8 = 3 * Ee * Ee / 8;
    f32_to_bf16_kernel<<<(n8 + 255) / 256, 256, 0, stream>>>(wqkv, wq16, n8);
  }
  {
    int n8 = Ee * Ee / 8;
    f32_to_bf16_kernel<<<(n8 + 255) / 256, 256, 0, stream>>>(wout, wo16, n8);
  }

  // QKV projection: [8192,1024] x [3072,1024]^T -> Q/K rows + V transposed
  gemm_bf16_wmma<0><<<dim3(3 * Ee / 64, M / 128), 256, 0, stream>>>(
      x16, wq16, bqkv, nullptr, qkbf, vT, (int)M, 3 * Ee, Ee);

  // Attention
  flash_attn_wmma<<<dim3(Ss / 128, Bb * Hh), 256, 0, stream>>>(qkbf, vT, mask,
                                                               ctx);

  // Output projection: [8192,1024] x [1024,1024]^T -> f32 out (+bias)
  gemm_bf16_wmma<1><<<dim3(Ee / 64, M / 128), 256, 0, stream>>>(
      ctx, wo16, bout, out, nullptr, nullptr, (int)M, Ee, Ee);
}
